// Linear3Bit_80453327389320
// MI455X (gfx1250) — matrix operand order repaired and checked by reading, not compiled, not run on hardware
//
#include <hip/hip_runtime.h>

// ---------------------------------------------------------------------------
// 3-bit block-quantized linear: out[M,N] = X[M,K] * W[N,K]^T + bias
// M = 16384, N = 4096, K = 4096, quant group = 32 along K.
//
// Phase 1a: dequantize W (3-bit packed -> bf16 [N,K]) into d_ws.
// Phase 1b: convert X fp32 -> bf16 into d_ws (if it fits).
// Phase 2 : 256x128 block-tile GEMM, 512 threads (16 wave32), wave tile 32x64,
//           v_wmma_f32_16x16x32_bf16, double-buffered LDS, and CDNA5
//           async global->LDS copies (ASYNCcnt) for the A and B tiles.
// MODE 2: W+X preconverted (A and B both async-copied; no VALU in loop)
// MODE 1: W preconverted    (B async; A reg-staged with packed bf16 cvt)
// MODE 0: fused dequant     (everything reg-staged)
// ---------------------------------------------------------------------------

typedef __attribute__((ext_vector_type(16))) __bf16 v16bf;
typedef __attribute__((ext_vector_type(2)))  __bf16 v2bf;
typedef __attribute__((ext_vector_type(8)))  float  v8f;

#define IN_F  4096
#define OUT_F 4096
#define BM 256
#define BN 128
#define BK 32
#define TPB 512
// LDS row stride in ushort units: 32 bf16 + 8 pad = 80 bytes (16B aligned)
#define LDA 40
#define LDB 40

__device__ __forceinline__ unsigned pkbf2(float a, float b) {
#if __has_builtin(__builtin_amdgcn_cvt_pk_bf16_f32)
  union { v2bf v; unsigned u; } r;
  r.v = __builtin_amdgcn_cvt_pk_bf16_f32(a, b);
  return r.u;
#else
  union { __bf16 h[2]; unsigned u; } r;
  r.h[0] = (__bf16)a;
  r.h[1] = (__bf16)b;
  return r.u;
#endif
}

union Frag {
  v16bf v;
  uint4 q[2];
};

// CDNA5 async global->LDS copy, 16B per lane, tracked by ASYNCcnt.
__device__ __forceinline__ void async_b128_to_lds(unsigned lds_byte_addr,
                                                  unsigned gbl_byte_off,
                                                  const void* gbl_base) {
  asm volatile("global_load_async_to_lds_b128 %0, %1, %2"
               :
               : "v"(lds_byte_addr), "v"(gbl_byte_off),
                 "s"((unsigned long long)(size_t)gbl_base)
               : "memory");
}

__device__ __forceinline__ void wait_asynccnt0() {
  asm volatile("s_wait_asynccnt 0x0" ::: "memory");
}

// --------------------------- Phase 1a: dequant W ----------------------------
__global__ __launch_bounds__(256)
void dequant3bit_kernel(const int*   __restrict__ wq,     // [n_groups,12]
                        const float* __restrict__ wnorm,  // [n_groups]
                        __bf16*      __restrict__ wbf) {  // [N*K] row-major
  const int t = blockIdx.x * 256 + threadIdx.x;
  const int g = t >> 1;
  const int h = t & 1;
  const int* q = wq + g * 12 + h * 6;
  const float norm  = wnorm[g];
  const float scale = norm * (2.0f / 7.0f);
  union { unsigned u[8]; uint4 q4[2]; } u;
  #pragma unroll
  for (int tr = 0; tr < 2; ++tr) {
    const int b0 = q[tr * 3 + 0];
    const int b1 = q[tr * 3 + 1];
    const int b2 = q[tr * 3 + 2];
    int v[8];
    v[0] =  b0       & 7;
    v[1] = (b0 >> 3) & 7;
    v[2] = ((b0 >> 6) & 3) | ((b1 & 1) << 2);
    v[3] = (b1 >> 1) & 7;
    v[4] = (b1 >> 4) & 7;
    v[5] = ((b1 >> 7) & 1) | ((b2 & 3) << 1);
    v[6] = (b2 >> 2) & 7;
    v[7] = (b2 >> 5) & 7;
    #pragma unroll
    for (int j = 0; j < 4; ++j)
      u.u[tr * 4 + j] = pkbf2(fmaf((float)v[2*j],   scale, -norm),
                              fmaf((float)v[2*j+1], scale, -norm));
  }
  uint4* dst = (uint4*)(wbf + (size_t)g * 32 + h * 16);
  dst[0] = u.q4[0];
  dst[1] = u.q4[1];
}

// --------------------------- Phase 1b: convert X ----------------------------
__global__ __launch_bounds__(256)
void cvtx_kernel(const float* __restrict__ x, __bf16* __restrict__ xbf) {
  const size_t i = ((size_t)blockIdx.x * 256 + threadIdx.x) * 8;
  const float4* p = (const float4*)(x + i);
  const float4 f0 = p[0], f1 = p[1];
  uint4 u;
  u.x = pkbf2(f0.x, f0.y);
  u.y = pkbf2(f0.z, f0.w);
  u.z = pkbf2(f1.x, f1.y);
  u.w = pkbf2(f1.z, f1.w);
  *(uint4*)(xbf + i) = u;
}

// --------------------------- Phase 2: GEMM ----------------------------------
template <int MODE>   // 0 = fused dequant, 1 = W predeq, 2 = W+X preconverted
__global__ __launch_bounds__(TPB)
void linear3bit_gemm(const float*  __restrict__ x,      // [M, K] fp32
                     const int*    __restrict__ wq,     // packed (MODE 0)
                     const float*  __restrict__ wnorm,  // (MODE 0)
                     const __bf16* __restrict__ wbf,    // [N, K] bf16 (MODE>=1)
                     const __bf16* __restrict__ xbf,    // [M, K] bf16 (MODE==2)
                     const float*  __restrict__ bias,   // [N]
                     float*        __restrict__ out) {  // [M, N] fp32
  __shared__ __align__(16) unsigned short Ash[2][BM * LDA];
  __shared__ __align__(16) unsigned short Bsh[2][BN * LDB];

  const int tid  = threadIdx.x;
  const int lane = tid & 31;
  const int wave = tid >> 5;        // 0..15
  const int wm   = wave >> 1;       // 0..7  (32 M-rows per wave)
  const int wn   = wave & 1;        // 0..1  (64 N-cols per wave)

  const int m0 = blockIdx.y * BM;
  const int n0 = blockIdx.x * BN;

  // staging: A -> 2 threads/row (32B each); B -> 4 threads/row (16B each)
  const int arow  = tid >> 1;       // 0..255
  const int ahalf = tid & 1;        // 0..1
  const int brow  = tid >> 2;       // 0..127
  const int bq    = tid & 3;        // 0..3

  // fragment addressing: contiguous k = 16*hic + s, same order for A and B
  const int ar   = lane & 15;
  const int hic  = lane >> 4;
  const int kb_a = hic * 16;
  const int kb_b = hic * 16;

  v8f acc[2][4];
  const v8f vzero = {0.f, 0.f, 0.f, 0.f, 0.f, 0.f, 0.f, 0.f};
  #pragma unroll
  for (int i = 0; i < 2; ++i)
    #pragma unroll
    for (int j = 0; j < 4; ++j)
      acc[i][j] = vzero;

  // A addressing
  const float* xrow = x + (size_t)(m0 + arow) * IN_F + ahalf * 16;
  const unsigned agbase  = ((unsigned)(m0 + arow) * IN_F + ahalf * 16) * 2u; // bf16 bytes
  const unsigned aldsoff = (unsigned)(arow * LDA + ahalf * 16) * 2u;
  // B addressing (byte offsets < 32MB fit u32)
  const unsigned bgbase  = ((unsigned)(n0 + brow) * IN_F + bq * 8) * 2u;
  const unsigned bldsoff = (unsigned)(brow * LDB + bq * 8) * 2u;
  const unsigned ash0 = (unsigned)(size_t)&Ash[0][0];
  const unsigned ash1 = (unsigned)(size_t)&Ash[1][0];
  const unsigned bsh0 = (unsigned)(size_t)&Bsh[0][0];
  const unsigned bsh1 = (unsigned)(size_t)&Bsh[1][0];

  // staged registers (MODE < 2)
  float4 xa[4];
  int    qb[3];
  float  nrm = 0.f;

  auto LOAD_A = [&](int k0) {
    const float4* xp = (const float4*)(xrow + k0);
    #pragma unroll
    for (int i = 0; i < 4; ++i) xa[i] = xp[i];
  };
  auto LOAD_B_FB = [&](int k0) {
    const int  g = (n0 + brow) * (IN_F / 32) + (k0 >> 5);
    const int* q = wq + g * 12 + bq * 3;
    qb[0] = q[0]; qb[1] = q[1]; qb[2] = q[2];
    nrm = wnorm[g];
  };
  auto STORE_A = [&](int buf) {
    uint4 p[2];
    #pragma unroll
    for (int i = 0; i < 2; ++i) {
      p[i].x = pkbf2(xa[2*i].x,   xa[2*i].y);
      p[i].y = pkbf2(xa[2*i].z,   xa[2*i].w);
      p[i].z = pkbf2(xa[2*i+1].x, xa[2*i+1].y);
      p[i].w = pkbf2(xa[2*i+1].z, xa[2*i+1].w);
    }
    uint4* ad = (uint4*)&Ash[buf][arow * LDA + ahalf * 16];
    ad[0] = p[0];
    ad[1] = p[1];
  };
  auto STORE_B_FB = [&](int buf) {
    const float scale = nrm * (2.0f / 7.0f);
    const int b0 = qb[0], b1 = qb[1], b2 = qb[2];
    int v[8];
    v[0] =  b0       & 7;
    v[1] = (b0 >> 3) & 7;
    v[2] = ((b0 >> 6) & 3) | ((b1 & 1) << 2);
    v[3] = (b1 >> 1) & 7;
    v[4] = (b1 >> 4) & 7;
    v[5] = ((b1 >> 7) & 1) | ((b2 & 3) << 1);
    v[6] = (b2 >> 2) & 7;
    v[7] = (b2 >> 5) & 7;
    uint4 u;
    u.x = pkbf2(fmaf((float)v[0], scale, -nrm), fmaf((float)v[1], scale, -nrm));
    u.y = pkbf2(fmaf((float)v[2], scale, -nrm), fmaf((float)v[3], scale, -nrm));
    u.z = pkbf2(fmaf((float)v[4], scale, -nrm), fmaf((float)v[5], scale, -nrm));
    u.w = pkbf2(fmaf((float)v[6], scale, -nrm), fmaf((float)v[7], scale, -nrm));
    *(uint4*)&Bsh[buf][brow * LDB + bq * 8] = u;
  };
  auto ASYNC_STAGE = [&](int dstbuf, int k0) {
    const unsigned kbytes = (unsigned)k0 * 2u;
    if (MODE >= 1)
      async_b128_to_lds((dstbuf ? bsh1 : bsh0) + bldsoff, bgbase + kbytes, wbf);
    if (MODE == 2) {
      const unsigned al = (dstbuf ? ash1 : ash0) + aldsoff;
      async_b128_to_lds(al,      agbase + kbytes,       xbf);
      async_b128_to_lds(al + 16, agbase + kbytes + 16u, xbf);
    }
  };

  // ---- prologue: stage K-step 0 ----
  if (MODE < 2) LOAD_A(0);
  if (MODE == 0) LOAD_B_FB(0);
  ASYNC_STAGE(0, 0);

  int buf = 0;
  for (int k0 = 0; k0 < IN_F; k0 += BK, buf ^= 1) {
    const bool has_next = (k0 + BK < IN_F);

    if (MODE < 2) {
      STORE_A(buf);
      if (MODE == 0) STORE_B_FB(buf);
      if (has_next) {
        LOAD_A(k0 + BK);
        if (MODE == 0) LOAD_B_FB(k0 + BK);
      }
    }
    if (MODE >= 1) wait_asynccnt0();   // async tiles for k0 resident in LDS[buf]
    __syncthreads();
    if (has_next) {
      // safe: every wave drained its ds reads of LDS[buf^1] (s_wait_dscnt
      // before its WMMAs) prior to signaling the barrier above.
      ASYNC_STAGE(buf ^ 1, k0 + BK);
    }

    Frag a[2], b[4];
    #pragma unroll
    for (int mi = 0; mi < 2; ++mi) {
      const unsigned short* p = &Ash[buf][(wm * 32 + mi * 16 + ar) * LDA];
      a[mi].q[0] = *(const uint4*)(p + kb_a);
      a[mi].q[1] = *(const uint4*)(p + 8 + kb_a);
    }
    #pragma unroll
    for (int ni = 0; ni < 4; ++ni) {
      const unsigned short* p = &Bsh[buf][(wn * 64 + ni * 16 + ar) * LDB + kb_b];
      b[ni].q[0] = ((const uint4*)p)[0];
      b[ni].q[1] = ((const uint4*)p)[1];
    }
    #pragma unroll
    for (int mi = 0; mi < 2; ++mi)
      #pragma unroll
      for (int ni = 0; ni < 4; ++ni)
        acc[mi][ni] = __builtin_amdgcn_wmma_f32_16x16x32_bf16(
            false, a[mi].v, false, b[ni].v, (short)0, acc[mi][ni], false, false);
  }

  // ---- epilogue: bias + fp32 store ----
  #pragma unroll
  for (int ni = 0; ni < 4; ++ni) {
    const int   n  = n0 + wn * 64 + ni * 16 + ar;
    const float bv = bias[n];
    #pragma unroll
    for (int mi = 0; mi < 2; ++mi) {
      #pragma unroll
      for (int r = 0; r < 8; ++r) {
        const int m = m0 + wm * 32 + mi * 16 + hic * 8 + r;
        out[(size_t)m * OUT_F + n] = acc[mi][ni][r] + bv;
      }
    }
  }
}

extern "C" void kernel_launch(void* const* d_in, const int* in_sizes, int n_in,
                              void* d_out, int out_size, void* d_ws, size_t ws_size,
                              hipStream_t stream) {
  const float* x     = (const float*)d_in[0];
  const int*   wq    = (const int*)d_in[1];
  const float* wnorm = (const float*)d_in[2];
  const float* bias  = (const float*)d_in[3];
  float*       out   = (float*)d_out;

  const int    M     = in_sizes[0] / IN_F;               // 16384
  dim3 grid(OUT_F / BN, M / BM);                         // (32, 64)
  const size_t needW = (size_t)OUT_F * IN_F * 2;         // 32 MB
  const size_t needX = (size_t)M * IN_F * 2;             // 128 MB

  if (ws_size >= needW + needX) {
    __bf16* wbf = (__bf16*)d_ws;
    __bf16* xbf = (__bf16*)((char*)d_ws + needW);
    dequant3bit_kernel<<<(OUT_F * IN_F / 32) * 2 / 256, 256, 0, stream>>>(wq, wnorm, wbf);
    cvtx_kernel<<<(int)((size_t)M * IN_F / 8 / 256), 256, 0, stream>>>(x, xbf);
    linear3bit_gemm<2><<<grid, TPB, 0, stream>>>(x, wq, wnorm, wbf, xbf, bias, out);
  } else if (ws_size >= needW) {
    __bf16* wbf = (__bf16*)d_ws;
    dequant3bit_kernel<<<(OUT_F * IN_F / 32) * 2 / 256, 256, 0, stream>>>(wq, wnorm, wbf);
    linear3bit_gemm<1><<<grid, TPB, 0, stream>>>(x, wq, wnorm, wbf, nullptr, bias, out);
  } else {
    linear3bit_gemm<0><<<grid, TPB, 0, stream>>>(x, wq, wnorm, nullptr, nullptr, bias, out);
  }
  (void)n_in; (void)out_size;
}
